// MPCN_86053964743375
// MI455X (gfx1250) — matrix operand order repaired and checked by reading, not compiled, not run on hardware
//
#include <hip/hip_runtime.h>
#include <hip/hip_bf16.h>
#include <math.h>

// Problem dims
#define NB 128     // batch
#define NR 30      // reviews
#define NL 60      // words per review
#define ND 300     // feature dim
#define NP 3       // pointers
#define DP 320     // padded feature dim (multiple of 32 for WMMA K)
#define RP 32      // padded review count
#define LP 64      // padded word count
#define WSZ (DP*DP)  // halfs per padded weight matrix

typedef __attribute__((ext_vector_type(16))) _Float16 v16h;
typedef __attribute__((ext_vector_type(8)))  _Float16 v8h;
typedef __attribute__((ext_vector_type(8)))  float    v8f;

// ---------------------------------------------------------------------------
// Fragment layouts (CDNA5 WMMA f16 16x16x32, wave32):
// A-frag: lane<16 holds row M=lane, K {0..7,16..23};
//         lane>=16 holds row M=lane-16, K {8..15,24..31}.
// B-frag: lane<16 holds col N=lane, K {0..7,16..23};
//         lane>=16 holds col N=lane-16, K {8..15,24..31}.
// D-frag: VGPR v, lane<16: (M=v, N=lane); lane>=16: (M=8+v, N=lane-16).
// ---------------------------------------------------------------------------
__device__ __forceinline__ v16h load_afrag(const _Float16* __restrict__ p) {
  v8h a0 = *(const v8h*)(p);
  v8h a1 = *(const v8h*)(p + 16);
  v16h a;
#pragma unroll
  for (int i = 0; i < 8; ++i) { a[i] = a0[i]; a[i+8] = a1[i]; }
  return a;
}
__device__ __forceinline__ v16h load_bfrag(const _Float16* __restrict__ p) {
  v8h b0 = *(const v8h*)(p);
  v8h b1 = *(const v8h*)(p + 16);
  v16h b;
#pragma unroll
  for (int i = 0; i < 8; ++i) { b[i] = b0[i]; b[i+8] = b1[i]; }
  return b;
}

// ---------------------------------------------------------------------------
// C[N,320] = A[N,320] * B[320,320]^T ; one wave per 16x64 strip (4 M-tiles).
// All 10 b128 loads of a k-step are issued before the 4 WMMAs so the
// scheduler can pipeline loads against matrix ops; A-frag reused 4x.
// ---------------------------------------------------------------------------
__global__ __launch_bounds__(128)
void wmma_gemm_nt4(const _Float16* __restrict__ A, const _Float16* __restrict__ B,
                   float* __restrict__ C, int Ntiles)
{
  const int Mgroups = DP / 64;                  // 5
  int wave = blockIdx.x * (blockDim.x >> 5) + (threadIdx.x >> 5);
  if (wave >= Ntiles * Mgroups) return;         // wave-uniform exit (EXEC stays full)
  int lane = threadIdx.x & 31;
  int tn = wave / Mgroups, tg = wave % Mgroups;
  int hi = lane >> 4;
  const _Float16* ap  = A + (size_t)(tn*16 + (lane & 15)) * DP + hi * 8;
  const _Float16* bp0 = B + (size_t)(tg*64 + (lane & 15)) * DP + hi * 8;
  v8f acc[4] = {v8f{}, v8f{}, v8f{}, v8f{}};
#pragma unroll 2
  for (int k = 0; k < DP; k += 32) {
    v16h a = load_afrag(ap + k);
    v16h bf[4];
#pragma unroll
    for (int t = 0; t < 4; ++t)
      bf[t] = load_bfrag(bp0 + (size_t)t * 16 * DP + k);
#pragma unroll
    for (int t = 0; t < 4; ++t)
      acc[t] = __builtin_amdgcn_wmma_f32_16x16x32_f16(
          false, a, false, bf[t], (short)0, acc[t], false, false);
  }
  int orow = tn*16 + hi*8;
#pragma unroll
  for (int t = 0; t < 4; ++t) {
    int ocol = tg*64 + t*16 + (lane & 15);
#pragma unroll
    for (int v = 0; v < 8; ++v)
      C[(size_t)(orow + v) * DP + ocol] = acc[t][v];
  }
}

// Batched bilinear scores: C[b] = A[b](T*16 x 320) * B[b]^T  (tiny; 1 tile/wave)
__global__ __launch_bounds__(32)
void wmma_gemm_nt_bat(const _Float16* __restrict__ A, const _Float16* __restrict__ B,
                      float* __restrict__ C, int strideA, int strideB,
                      int ldc, int strideC)
{
  int b = blockIdx.x, tn = blockIdx.y, tm = blockIdx.z;
  int lane = threadIdx.x & 31;
  int hi = lane >> 4;
  const _Float16* ap = A + (size_t)b*strideA + (size_t)(tn*16 + (lane & 15))*DP + hi*8;
  const _Float16* bp = B + (size_t)b*strideB + (size_t)(tm*16 + (lane & 15))*DP + hi*8;
  v8f acc = {};
#pragma unroll 4
  for (int k = 0; k < DP; k += 32) {
    v16h a = load_afrag(ap + k);
    v16h bf = load_bfrag(bp + k);
    acc = __builtin_amdgcn_wmma_f32_16x16x32_f16(
        false, a, false, bf, (short)0, acc, false, false);
  }
  float* cp = C + (size_t)b * strideC;
  int orow = tn*16 + hi*8, ocol = tm*16 + (lane & 15);
#pragma unroll
  for (int v = 0; v < 8; ++v)
    cp[(size_t)(orow + v) * ldc + ocol] = acc[v];
}

// ---------------------------------------------------------------------------
// Weight padding 300x300 f32 -> 320x320 f16 (optional transpose for M matrices)
// ---------------------------------------------------------------------------
__global__ void pad_weight_h(const float* __restrict__ src, _Float16* __restrict__ dst,
                             int transpose)
{
  int i = blockIdx.x * 256 + threadIdx.x;
  if (i >= WSZ) return;
  int k = i % DP, m = i / DP;
  float v = 0.f;
  if (m < ND && k < ND) v = transpose ? src[k*ND + m] : src[m*ND + k];
  dst[i] = (_Float16)v;
}

// ---------------------------------------------------------------------------
// Sum word embeddings of a review -> padded f16 [B,RP,DP]
// ---------------------------------------------------------------------------
__global__ void embed_sum_h(const int* __restrict__ reviews, const float* __restrict__ emb,
                            _Float16* __restrict__ out)
{
  int br = blockIdx.x;            // b*RP + r
  int r = br & (RP - 1), b = br >> 5;
  int d = threadIdx.x;            // 0..319
  __shared__ int toks[NL];
  if (d < NL && r < NR) toks[d] = reviews[((size_t)b*NR + r)*NL + d];
  __syncthreads();
  float acc = 0.f;
  if (r < NR && d < ND)
    for (int l = 0; l < NL; ++l) acc += emb[(size_t)toks[l]*ND + d];
  out[(size_t)br*DP + d] = (_Float16)((r < NR && d < ND) ? acc : 0.f);
}

// gate: out = sigmoid(z1+b1)*tanh(z2+b2) -> f16 padded
__global__ void gate_epi(const float* __restrict__ z1, const float* __restrict__ z2,
                         const float* __restrict__ b1, const float* __restrict__ b2,
                         _Float16* __restrict__ out, int rowMask, int rowValid, int total)
{
  int i = blockIdx.x * 256 + threadIdx.x;
  if (i >= total) return;
  int d = i % DP, row = i / DP, r = row & rowMask;
  float v = 0.f;
  if (r < rowValid && d < ND) {
    float a = z1[i] + b1[d];
    float t = z2[i] + b2[d];
    v = (1.f / (1.f + __expf(-a))) * tanhf(t);
  }
  out[i] = (_Float16)v;
}

// out = z (+ bias) -> f16 padded, zeroing invalid rows/cols
__global__ void bias_epi(const float* __restrict__ z, const float* __restrict__ bias,
                         _Float16* __restrict__ out, int rowMask, int rowValid, int total)
{
  int i = blockIdx.x * 256 + threadIdx.x;
  if (i >= total) return;
  int d = i % DP, row = i / DP, r = row & rowMask;
  float v = 0.f;
  if (r < rowValid && d < ND) v = z[i] + (bias ? bias[d] : 0.f);
  out[i] = (_Float16)v;
}

// ---------------------------------------------------------------------------
// Review-level: max-pool scores + deterministic gumbel + argmax
// ---------------------------------------------------------------------------
__device__ __forceinline__ unsigned mix32(unsigned x) {
  x ^= x >> 16; x *= 0x7feb352dU; x ^= x >> 15; x *= 0x846ca68bU; x ^= x >> 16;
  return x;
}
__device__ __forceinline__ float gumbel_det(unsigned seed) {
  unsigned h = mix32(seed * 0x9e3779b9U + 0x85ebca6bU);
  float u = ((float)h + 0.5f) * (1.0f / 4294967296.0f);
  return -__logf(-__logf(u));
}

__global__ void review_argmax(const float* __restrict__ S, int* __restrict__ idxu,
                              int* __restrict__ idxi, int p)
{
  int b = blockIdx.x, r = threadIdx.x;   // 32 threads
  __shared__ float su[RP], si[RP];
  float mu = -1e30f, mi = -1e30f;
  const float* Sb = S + (size_t)b * RP * RP;
  if (r < NR) {
    for (int s = 0; s < NR; ++s) {
      mu = fmaxf(mu, Sb[r*RP + s]);
      mi = fmaxf(mi, Sb[s*RP + r]);
    }
    mu += gumbel_det((unsigned)((2*p)     * 65536 + b * 64 + r));
    mi += gumbel_det((unsigned)((2*p + 1) * 65536 + b * 64 + r));
  }
  su[r] = (r < NR) ? mu : -1e30f;
  si[r] = (r < NR) ? mi : -1e30f;
  __syncthreads();
  if (r == 0) {
    int bu = 0, bi = 0; float xu = su[0], xi = si[0];
    for (int t = 1; t < NR; ++t) {
      if (su[t] > xu) { xu = su[t]; bu = t; }
      if (si[t] > xi) { xi = si[t]; bi = t; }
    }
    idxu[b] = bu; idxi[b] = bi;
  }
}

// Gather word embeddings of the selected review -> f16 [B,LP,DP]
__global__ void gather_words_h(const int* __restrict__ reviews, const float* __restrict__ emb,
                               const int* __restrict__ idx, _Float16* __restrict__ out)
{
  int bl = blockIdx.x;           // b*LP + l
  int l = bl & (LP - 1), b = bl >> 6;
  int d = threadIdx.x;
  float v = 0.f;
  if (l < NL && d < ND) {
    int tok = reviews[((size_t)b*NR + idx[b])*NL + l];
    v = emb[(size_t)tok*ND + d];
  }
  out[(size_t)bl*DP + d] = (_Float16)v;
}

// Word-level: mean-pool + softmax over valid words (both directions)
__global__ void word_attn(const float* __restrict__ S2, float* __restrict__ pu,
                          float* __restrict__ pi)
{
  int b = blockIdx.x, l = threadIdx.x;   // 64 threads
  __shared__ float ru[LP], ri[LP];
  const float* Sb = S2 + (size_t)b * LP * LP;
  float su = 0.f, sc = 0.f;
  if (l < NL) {
    for (int j = 0; j < NL; ++j) { su += Sb[l*LP + j]; sc += Sb[j*LP + l]; }
  }
  ru[l] = (l < NL) ? su * (1.f/NL) : -1e30f;
  ri[l] = (l < NL) ? sc * (1.f/NL) : -1e30f;
  __syncthreads();
  float mu = -1e30f, mi = -1e30f;
  for (int j = 0; j < NL; ++j) { mu = fmaxf(mu, ru[j]); mi = fmaxf(mi, ri[j]); }
  float eu = (l < NL) ? __expf(ru[l] - mu) : 0.f;
  float ei = (l < NL) ? __expf(ri[l] - mi) : 0.f;
  __shared__ float xu[LP], xi[LP];
  xu[l] = eu; xi[l] = ei;
  __syncthreads();
  float du = 0.f, di = 0.f;
  for (int j = 0; j < NL; ++j) { du += xu[j]; di += xi[j]; }
  pu[(size_t)b*LP + l] = (l < NL) ? eu / du : 0.f;
  pi[(size_t)b*LP + l] = (l < NL) ? ei / di : 0.f;
}

// rep[b, p*300 + d] = sum_l p[b,l] * emb[token(b, idx[b], l), d]  (full f32)
__global__ void weighted_rep(const int* __restrict__ reviews, const float* __restrict__ emb,
                             const int* __restrict__ idx, const float* __restrict__ pv,
                             float* __restrict__ rep, int p)
{
  int b = blockIdx.x;
  int d = threadIdx.x;     // 320
  __shared__ float w[LP];
  __shared__ int  toks[LP];
  if (d < LP) {
    w[d] = pv[(size_t)b*LP + d];
    toks[d] = (d < NL) ? reviews[((size_t)b*NR + idx[b])*NL + d] : 0;
  }
  __syncthreads();
  if (d < ND) {
    float acc = 0.f;
    for (int l = 0; l < NL; ++l) acc += w[l] * emb[(size_t)toks[l]*ND + d];
    rep[(size_t)b*(NP*ND) + p*ND + d] = acc;
  }
}

// ---------------------------------------------------------------------------
// Factorization machine head
// ---------------------------------------------------------------------------
__device__ __forceinline__ float block_reduce256(float v, float* lds) {
  for (int o = 16; o > 0; o >>= 1) v += __shfl_down(v, o, 32);
  int wid = threadIdx.x >> 5;
  if ((threadIdx.x & 31) == 0) lds[wid] = v;
  __syncthreads();
  float r = 0.f;
  if (threadIdx.x == 0) for (int w = 0; w < 8; ++w) r += lds[w];
  __syncthreads();
  return r;   // valid on thread 0
}

__global__ __launch_bounds__(256)
void fm_head(const float* __restrict__ urep, const float* __restrict__ irep,
             const float* __restrict__ w0, const float* __restrict__ fw,
             const float* __restrict__ fv, float* __restrict__ out)
{
  const int JD = 2 * NP * ND;   // 1800
  int b = blockIdx.x, t = threadIdx.x;
  float lin = 0.f, sv[10], s2v[10];
#pragma unroll
  for (int f = 0; f < 10; ++f) { sv[f] = 0.f; s2v[f] = 0.f; }
  for (int j = t; j < JD; j += 256) {
    float x = (j < NP*ND) ? urep[(size_t)b*(NP*ND) + j]
                          : irep[(size_t)b*(NP*ND) + (j - NP*ND)];
    lin += fw[j] * x;
#pragma unroll
    for (int f = 0; f < 10; ++f) {
      float vv = fv[(size_t)j*10 + f];
      sv[f]  += x * vv;
      s2v[f] += x * x * vv * vv;
    }
  }
  __shared__ float lds[8];
  float L = block_reduce256(lin, lds);
  float acc = 0.f;
  for (int f = 0; f < 10; ++f) {
    float a = block_reduce256(sv[f],  lds);
    float c = block_reduce256(s2v[f], lds);
    if (t == 0) acc += a * a - c;
  }
  if (t == 0) out[b] = w0[0] + L + 0.5f * acc;
}

// ---------------------------------------------------------------------------
// Host orchestration
// ---------------------------------------------------------------------------
static inline int ceil_div(int a, int b) { return (a + b - 1) / b; }

extern "C" void kernel_launch(void* const* d_in, const int* in_sizes, int n_in,
                              void* d_out, int out_size, void* d_ws, size_t ws_size,
                              hipStream_t stream)
{
  const int*   user_reviews = (const int*)  d_in[0];
  const int*   item_reviews = (const int*)  d_in[1];
  const float* user_emb     = (const float*)d_in[2];
  const float* item_emb     = (const float*)d_in[3];
  const float* g1_w = (const float*)d_in[4];
  const float* g1_b = (const float*)d_in[5];
  const float* g2_w = (const float*)d_in[6];
  const float* g2_b = (const float*)d_in[7];
  const float* Mr   = (const float*)d_in[8];
  const float* Wur  = (const float*)d_in[9];
  const float* bur  = (const float*)d_in[10];
  const float* Wir  = (const float*)d_in[11];
  const float* bir  = (const float*)d_in[12];
  const float* Mw   = (const float*)d_in[13];
  const float* Wuw  = (const float*)d_in[14];
  const float* buw  = (const float*)d_in[15];
  const float* Wiw  = (const float*)d_in[16];
  const float* biw  = (const float*)d_in[17];
  const float* fm_w0 = (const float*)d_in[18];
  const float* fm_w  = (const float*)d_in[19];
  const float* fm_v  = (const float*)d_in[20];
  float* out = (float*)d_out;

  // workspace carve-out
  size_t off = 0;
  auto carve = [&](size_t bytes) -> char* {
    char* p = (char*)d_ws + off;
    off += (bytes + 255) & ~(size_t)255;
    return p;
  };
  _Float16* WH   = (_Float16*)carve((size_t)20 * WSZ * 2);     // padded weights
  _Float16* su_h = (_Float16*)carve((size_t)NB*RP*DP * 2);
  _Float16* si_h = (_Float16*)carve((size_t)NB*RP*DP * 2);
  _Float16* urev = (_Float16*)carve((size_t)NB*RP*DP * 2);
  _Float16* irev = (_Float16*)carve((size_t)NB*RP*DP * 2);
  _Float16* H1 = (_Float16*)carve((size_t)NB*LP*DP * 2);
  _Float16* H2 = (_Float16*)carve((size_t)NB*LP*DP * 2);
  _Float16* H3 = (_Float16*)carve((size_t)NB*LP*DP * 2);
  _Float16* H4 = (_Float16*)carve((size_t)NB*LP*DP * 2);
  _Float16* H5 = (_Float16*)carve((size_t)NB*LP*DP * 2);
  float* z1 = (float*)carve((size_t)NB*LP*DP * 4);
  float* z2 = (float*)carve((size_t)NB*LP*DP * 4);
  float* S  = (float*)carve((size_t)NB*LP*LP * 4);
  float* pu = (float*)carve((size_t)NB*LP * 4);
  float* pi = (float*)carve((size_t)NB*LP * 4);
  int* idxu = (int*)carve((size_t)NB * 4);
  int* idxi = (int*)carve((size_t)NB * 4);
  float* urep = (float*)carve((size_t)NB*NP*ND * 4);
  float* irep = (float*)carve((size_t)NB*NP*ND * 4);
  (void)ws_size; (void)in_sizes; (void)n_in; (void)out_size;

  // weight slots in WH
  const int I_G1 = 0, I_G2 = 1, I_WUR = 2, I_WIR = 5, I_MRT = 8,
            I_WUW = 11, I_WIW = 14, I_MWT = 17;
  const int padBlocks = ceil_div(WSZ, 256);
  hipLaunchKernelGGL(pad_weight_h, dim3(padBlocks), dim3(256), 0, stream, g1_w, WH + (size_t)I_G1*WSZ, 0);
  hipLaunchKernelGGL(pad_weight_h, dim3(padBlocks), dim3(256), 0, stream, g2_w, WH + (size_t)I_G2*WSZ, 0);
  for (int p = 0; p < NP; ++p) {
    hipLaunchKernelGGL(pad_weight_h, dim3(padBlocks), dim3(256), 0, stream, Wur + (size_t)p*ND*ND, WH + (size_t)(I_WUR+p)*WSZ, 0);
    hipLaunchKernelGGL(pad_weight_h, dim3(padBlocks), dim3(256), 0, stream, Wir + (size_t)p*ND*ND, WH + (size_t)(I_WIR+p)*WSZ, 0);
    hipLaunchKernelGGL(pad_weight_h, dim3(padBlocks), dim3(256), 0, stream, Mr  + (size_t)p*ND*ND, WH + (size_t)(I_MRT+p)*WSZ, 1);
    hipLaunchKernelGGL(pad_weight_h, dim3(padBlocks), dim3(256), 0, stream, Wuw + (size_t)p*ND*ND, WH + (size_t)(I_WUW+p)*WSZ, 0);
    hipLaunchKernelGGL(pad_weight_h, dim3(padBlocks), dim3(256), 0, stream, Wiw + (size_t)p*ND*ND, WH + (size_t)(I_WIW+p)*WSZ, 0);
    hipLaunchKernelGGL(pad_weight_h, dim3(padBlocks), dim3(256), 0, stream, Mw  + (size_t)p*ND*ND, WH + (size_t)(I_MWT+p)*WSZ, 1);
  }

  // 1) per-review word-sum embeddings (padded f16)
  hipLaunchKernelGGL(embed_sum_h, dim3(NB*RP), dim3(DP), 0, stream, user_reviews, user_emb, su_h);
  hipLaunchKernelGGL(embed_sum_h, dim3(NB*RP), dim3(DP), 0, stream, item_reviews, item_emb, si_h);

  // GEMM launch helper: wave computes 16x64 strip; Mgroups = 320/64 = 5
  const int NtR = NB*RP/16, NtW = NB*LP/16, Mg = DP/64;   // 256, 512, 5
  auto gemm = [&](const _Float16* A, const _Float16* Bm, float* C, int Ntiles) {
    int waves = Ntiles * Mg;
    hipLaunchKernelGGL(wmma_gemm_nt4, dim3(ceil_div(waves, 4)), dim3(128), 0, stream,
                       A, Bm, C, Ntiles);
  };
  const int totR = NB*RP*DP, totW = NB*LP*DP;

  // 2) review gate -> urev/irev (f16)
  gemm(su_h, WH + (size_t)I_G1*WSZ, z1, NtR);
  gemm(su_h, WH + (size_t)I_G2*WSZ, z2, NtR);
  hipLaunchKernelGGL(gate_epi, dim3(ceil_div(totR,256)), dim3(256), 0, stream,
                     z1, z2, g1_b, g2_b, urev, RP-1, NR, totR);
  gemm(si_h, WH + (size_t)I_G1*WSZ, z1, NtR);
  gemm(si_h, WH + (size_t)I_G2*WSZ, z2, NtR);
  hipLaunchKernelGGL(gate_epi, dim3(ceil_div(totR,256)), dim3(256), 0, stream,
                     z1, z2, g1_b, g2_b, irev, RP-1, NR, totR);

  // 3) per-pointer co-attention
  for (int p = 0; p < NP; ++p) {
    // review-level: u' = urev@Wur^T+bur ; i' = irev@Wir^T+bir ; t = u'@Mr
    gemm(urev, WH + (size_t)(I_WUR+p)*WSZ, z1, NtR);
    hipLaunchKernelGGL(bias_epi, dim3(ceil_div(totR,256)), dim3(256), 0, stream,
                       z1, bur + (size_t)p*ND, H1, RP-1, NR, totR);
    gemm(irev, WH + (size_t)(I_WIR+p)*WSZ, z1, NtR);
    hipLaunchKernelGGL(bias_epi, dim3(ceil_div(totR,256)), dim3(256), 0, stream,
                       z1, bir + (size_t)p*ND, H3, RP-1, NR, totR);
    gemm(H1, WH + (size_t)(I_MRT+p)*WSZ, z1, NtR);
    hipLaunchKernelGGL(bias_epi, dim3(ceil_div(totR,256)), dim3(256), 0, stream,
                       z1, (const float*)nullptr, H2, RP-1, NR, totR);
    // scores S[b] = t[b] * i'[b]^T   [B,32,32]
    hipLaunchKernelGGL(wmma_gemm_nt_bat, dim3(NB, RP/16, RP/16), dim3(32), 0, stream,
                       H2, H3, S, RP*DP, RP*DP, RP, RP*RP);
    hipLaunchKernelGGL(review_argmax, dim3(NB), dim3(RP), 0, stream, S, idxu, idxi, p);

    // word-level: gather selected reviews' words
    hipLaunchKernelGGL(gather_words_h, dim3(NB*LP), dim3(DP), 0, stream,
                       user_reviews, user_emb, idxu, H4);
    hipLaunchKernelGGL(gather_words_h, dim3(NB*LP), dim3(DP), 0, stream,
                       item_reviews, item_emb, idxi, H5);
    // u'w = H4@Wuw^T+buw -> H1 ; t_w = H1@Mw -> H2 ; i'w = H5@Wiw^T+biw -> H3
    gemm(H4, WH + (size_t)(I_WUW+p)*WSZ, z1, NtW);
    hipLaunchKernelGGL(bias_epi, dim3(ceil_div(totW,256)), dim3(256), 0, stream,
                       z1, buw + (size_t)p*ND, H1, LP-1, NL, totW);
    gemm(H1, WH + (size_t)(I_MWT+p)*WSZ, z1, NtW);
    hipLaunchKernelGGL(bias_epi, dim3(ceil_div(totW,256)), dim3(256), 0, stream,
                       z1, (const float*)nullptr, H2, LP-1, NL, totW);
    gemm(H5, WH + (size_t)(I_WIW+p)*WSZ, z1, NtW);
    hipLaunchKernelGGL(bias_epi, dim3(ceil_div(totW,256)), dim3(256), 0, stream,
                       z1, biw + (size_t)p*ND, H3, LP-1, NL, totW);
    // S2[b] = t_w[b] * i'w[b]^T   [B,64,64]
    hipLaunchKernelGGL(wmma_gemm_nt_bat, dim3(NB, LP/16, LP/16), dim3(32), 0, stream,
                       H2, H3, S, LP*DP, LP*DP, LP, LP*LP);
    hipLaunchKernelGGL(word_attn, dim3(NB), dim3(LP), 0, stream, S, pu, pi);
    // weighted word reps (full f32 via re-gather)
    hipLaunchKernelGGL(weighted_rep, dim3(NB), dim3(DP), 0, stream,
                       user_reviews, user_emb, idxu, pu, urep, p);
    hipLaunchKernelGGL(weighted_rep, dim3(NB), dim3(DP), 0, stream,
                       item_reviews, item_emb, idxi, pi, irep, p);
  }

  // 4) FM head
  hipLaunchKernelGGL(fm_head, dim3(NB), dim3(256), 0, stream,
                     urep, irep, fm_w0, fm_w, fm_v, out);
}
